// GQAWithKVCache_74036646249077
// MI455X (gfx1250) — matrix operand order repaired and checked by reading, not compiled, not run on hardware
//
#include <hip/hip_runtime.h>
#include <hip/hip_bf16.h>

// ---------------- problem constants (from reference) ----------------
#define BB     64
#define HQ     32
#define HKV    8
#define HD     64
#define DD     2048
#define MAXKV  4096
#define GG     (HQ / HKV)          // 4
#define NQKV   (HQ*HD + 2*HKV*HD)  // 3072
#define TILE_T 256                 // K/V rows staged per attention tile
#define LDSTR  66                  // padded LDS row stride (dwords), conflict-free

typedef __bf16 bf16_t;
typedef __attribute__((ext_vector_type(16))) __bf16 v16bf;
typedef __attribute__((ext_vector_type(8)))  float  v8f;

// =====================================================================
// RMSNorm: h = x * rsqrt(mean(x^2)+eps) * w   -> bf16
// =====================================================================
__global__ __launch_bounds__(256)
void rmsnorm_kernel(const float* __restrict__ x, const float* __restrict__ w,
                    bf16_t* __restrict__ h) {
  const int b = blockIdx.x, tid = threadIdx.x;
  __shared__ float red[256];
  const float* xr = x + (size_t)b * DD;
  float ss = 0.f;
  for (int i = tid; i < DD; i += 256) { float v = xr[i]; ss += v * v; }
  red[tid] = ss; __syncthreads();
  for (int s = 128; s > 0; s >>= 1) {
    if (tid < s) red[tid] += red[tid + s];
    __syncthreads();
  }
  const float scale = rsqrtf(red[0] / (float)DD + 1e-9f);
  for (int i = tid; i < DD; i += 256)
    h[(size_t)b * DD + i] = (bf16_t)(xr[i] * scale * w[i]);
}

// =====================================================================
// One 16x16 output tile via v_wmma_f32_16x16x32_bf16, K-loop.
// A: bf16 MxK row-major. W: f32 KxN row-major (converted to bf16 in regs).
// Fragment layouts per CDNA5 ISA 7.12.2:
//   A 16x32: lane<16 row m=lane holds K {0..7,16..23}; lane>=16 same row, +8
//   B 32x16: lane n=lane&15; halves i: K same as A (i<8 ? i : 8+i, +8 if lane>=16)
//   C/D:     vgpr j -> row m = j + 8*(lane>=16), col n = lane&15
// =====================================================================
__device__ inline v8f wmma_tile_bf16(const bf16_t* __restrict__ A, int lda,
                                     const float* __restrict__ W, int ldw,
                                     int c0, int K) {
  const int lane = threadIdx.x & 31;
  const int nm = lane & 15;
  const int hi = lane >> 4;
  v8f acc = {};
  for (int k0 = 0; k0 < K; k0 += 32) {
    v16bf a, bm;
#pragma unroll
    for (int i = 0; i < 16; ++i) {
      const int ka = (i < 8 ? i : 8 + i) + (hi ? 8 : 0);
      a[i] = A[(size_t)nm * lda + k0 + ka];
      const int kb = (i < 8 ? i : 8 + i) + (hi ? 8 : 0);
      bm[i] = (bf16_t)W[(size_t)(k0 + kb) * ldw + c0 + nm];
    }
    acc = __builtin_amdgcn_wmma_f32_16x16x32_bf16(false, a, false, bm,
                                                  (short)0, acc, false, false);
  }
  return acc;
}

// QKV = h @ [Wq | Wk | Wv]  -> qkv (B x 3072, f32)
__global__ __launch_bounds__(32)
void qkv_gemm_kernel(const bf16_t* __restrict__ h,
                     const float* __restrict__ Wq, const float* __restrict__ Wk,
                     const float* __restrict__ Wv, float* __restrict__ qkv) {
  const int n0 = blockIdx.x * 16, m0 = blockIdx.y * 16;
  const float* W; int ldw, c0;
  if (n0 < HQ * HD)               { W = Wq; ldw = HQ * HD;  c0 = n0; }
  else if (n0 < HQ*HD + HKV*HD)   { W = Wk; ldw = HKV * HD; c0 = n0 - HQ*HD; }
  else                            { W = Wv; ldw = HKV * HD; c0 = n0 - HQ*HD - HKV*HD; }
  v8f acc = wmma_tile_bf16(h + (size_t)m0 * DD, DD, W, ldw, c0, DD);
  const int lane = threadIdx.x & 31, nm = lane & 15, hi = lane >> 4;
#pragma unroll
  for (int j = 0; j < 8; ++j)
    qkv[(size_t)(m0 + hi * 8 + j) * NQKV + n0 + nm] = acc[j];
}

// out = x + o @ Wo
__global__ __launch_bounds__(32)
void oproj_gemm_kernel(const bf16_t* __restrict__ oB, const float* __restrict__ Wo,
                       const float* __restrict__ x, float* __restrict__ out) {
  const int n0 = blockIdx.x * 16, m0 = blockIdx.y * 16;
  v8f acc = wmma_tile_bf16(oB + (size_t)m0 * DD, DD, Wo, DD, n0, DD);
  const int lane = threadIdx.x & 31, nm = lane & 15, hi = lane >> 4;
#pragma unroll
  for (int j = 0; j < 8; ++j) {
    const size_t r = (size_t)(m0 + hi * 8 + j) * DD + n0 + nm;
    out[r] = x[r] + acc[j];
  }
}

// =====================================================================
// RoPE in place on q (HQ heads) and k (HKV heads) inside qkv buffer.
// =====================================================================
__global__ __launch_bounds__(32)
void rope_kernel(float* __restrict__ qkv, const int* __restrict__ ctx_lens) {
  const int hh = blockIdx.x, b = blockIdx.y, i = threadIdx.x; // i in [0,32)
  float* base = qkv + (size_t)b * NQKV +
                (hh < HQ ? hh * HD : HQ * HD + (hh - HQ) * HD);
  const float pos = (float)ctx_lens[b];
  const float inv = powf(10000.f, -(float)i / 32.f);
  const float ang = pos * inv;
  const float c = cosf(ang), s = sinf(ang);
  const float x1 = base[i], x2 = base[i + 32];
  base[i]      = x1 * c - x2 * s;
  base[i + 32] = x1 * s + x2 * c;
}

// =====================================================================
// Flash-decode attention. Block = (kv head, batch). 8 waves, lane-per-t.
// K/V tiles staged to LDS by the Tensor Data Mover (row-padded to 66 dw).
// New-token (k,v) handled as a 9th softmax partial (caches are read-only).
// =====================================================================
__global__ __launch_bounds__(256)
void attn_kernel(const float* __restrict__ qkv,
                 const float* __restrict__ cache_k,
                 const float* __restrict__ cache_v,
                 const int* __restrict__ ctx_lens,
                 bf16_t* __restrict__ o_out) {
  const int h = blockIdx.x, b = blockIdx.y;
  const int tid = threadIdx.x, lane = tid & 31, w = tid >> 5;

  __shared__ float q_s[GG][HD];
  __shared__ float Ks[TILE_T * LDSTR];
  __shared__ float Vs[TILE_T * LDSTR];
  __shared__ float red_m[9][GG];
  __shared__ float red_l[9][GG];
  __shared__ float red_o[9][GG][HD];

  const float* qb = qkv + (size_t)b * NQKV;
  const float* kn = qb + HQ * HD + h * HD;            // roped new k
  const float* vn = qb + HQ * HD + HKV * HD + h * HD; // new v
  const float* Kc = cache_k + ((size_t)b * HKV + h) * (size_t)MAXKV * HD;
  const float* Vc = cache_v + ((size_t)b * HKV + h) * (size_t)MAXKV * HD;
  const int L0 = ctx_lens[b];

  for (int i = tid; i < GG * HD; i += 256) {
    const int g = i >> 6, d = i & 63;
    q_s[g][d] = qb[(h * GG + g) * HD + d] * 0.125f; // 1/sqrt(HD)
  }
  __syncthreads();

  float m_g[GG], l_g[GG], o0[GG], o1[GG];
#pragma unroll
  for (int g = 0; g < GG; ++g) { m_g[g] = -__builtin_inff(); l_g[g] = 0.f; o0[g] = 0.f; o1[g] = 0.f; }

  for (int tile = 0; tile < L0; tile += TILE_T) {
    const unsigned rows = (unsigned)(L0 - tile); // TDM clamps, OOB reads -> 0
#if defined(__has_builtin) && __has_builtin(__builtin_amdgcn_tensor_load_to_lds)
    if (w == 0) {
      typedef unsigned int v4u __attribute__((ext_vector_type(4)));
      typedef int v4i __attribute__((ext_vector_type(4)));
      typedef int v8i __attribute__((ext_vector_type(8)));
      // D# group1: data_size=4B, pad_enable, pad every 64 dw by 2 dw -> stride 66
      const unsigned g1d0 = (2u << 16) | (1u << 20) | (5u << 22) | (1u << 25);
      const unsigned g1d1 = (unsigned)HD << 16;                    // tensor_dim0=64
      const unsigned g1d2 = (rows & 0xFFFFu) << 16;                // tensor_dim1 lo
      const unsigned g1d3 = ((rows >> 16) & 0xFFFFu) | ((unsigned)HD << 16); // dim1 hi | tile_dim0
      const unsigned g1d4 = (unsigned)TILE_T;                      // tile_dim1
      const unsigned g1d5 = (unsigned)HD;                          // dim0_stride lo
      const v8i g1 = {(int)g1d0, (int)g1d1, (int)g1d2, (int)g1d3,
                      (int)g1d4, (int)g1d5, 0, 0};
      const v4i z4 = {0, 0, 0, 0};
#pragma unroll
      for (int kv = 0; kv < 2; ++kv) {
        const unsigned long long ga =
            (unsigned long long)(kv ? (Vc + (size_t)tile * HD) : (Kc + (size_t)tile * HD));
        const unsigned lds = (unsigned)(unsigned long long)(kv ? (void*)Vs : (void*)Ks);
        const v4u g0 = {1u, lds, (unsigned)ga,
                        (unsigned)((ga >> 32) & 0x1FFFFFFull) | (2u << 30)};
#if __clang_major__ >= 23
        const v8i z8 = {0, 0, 0, 0, 0, 0, 0, 0};
        __builtin_amdgcn_tensor_load_to_lds(g0, g1, z4, z4, z8, 0);
#else
        __builtin_amdgcn_tensor_load_to_lds(g0, g1, z4, z4, 0);
#endif
      }
      __builtin_amdgcn_s_wait_tensorcnt(0);
    }
#else
    // Fallback: cooperative synchronous staging (coalesced, zero-filled)
    for (int e = tid; e < TILE_T * HD; e += 256) {
      const int tr = e >> 6, d = e & 63;
      const int t = tile + tr;
      Ks[tr * LDSTR + d] = ((unsigned)tr < rows) ? Kc[(size_t)t * HD + d] : 0.f;
      Vs[tr * LDSTR + d] = ((unsigned)tr < rows) ? Vc[(size_t)t * HD + d] : 0.f;
    }
#endif
    __syncthreads();

    const int tl = w * 32 + lane;          // row inside tile, lane-per-token
    const bool valid = (unsigned)tl < rows;
    float s[GG] = {0.f, 0.f, 0.f, 0.f};
    const int krow = tl * LDSTR;
#pragma unroll 4
    for (int d = 0; d < HD; d += 2) {
      const float2 kv = *(const float2*)&Ks[krow + d];
#pragma unroll
      for (int g = 0; g < GG; ++g) {
        const float2 qv = *(const float2*)&q_s[g][d];
        s[g] += qv.x * kv.x + qv.y * kv.y;
      }
    }
    float pcur[GG];
#pragma unroll
    for (int g = 0; g < GG; ++g) {
      float sv = valid ? s[g] : -__builtin_inff();
      float smax = sv;
      for (int off = 16; off > 0; off >>= 1)
        smax = fmaxf(smax, __shfl_xor(smax, off, 32));
      if (smax < -1e37f) { pcur[g] = 0.f; continue; }  // whole slice masked
      const float mnew  = fmaxf(m_g[g], smax);
      const float alpha = __expf(m_g[g] - mnew);
      const float p     = valid ? __expf(sv - mnew) : 0.f;
      float psum = p;
      for (int off = 16; off > 0; off >>= 1) psum += __shfl_xor(psum, off, 32);
      l_g[g] = l_g[g] * alpha + psum;
      o0[g] *= alpha; o1[g] *= alpha;
      m_g[g] = mnew;
      pcur[g] = p;
    }
    // o += P^T V  (broadcast p across wave, lane owns d=lane, d=lane+32)
    for (int tt = 0; tt < 32; ++tt) {
      const int vr = (w * 32 + tt) * LDSTR;
      const float v0 = Vs[vr + lane], v1 = Vs[vr + 32 + lane];
#pragma unroll
      for (int g = 0; g < GG; ++g) {
        const float pg = __shfl(pcur[g], tt, 32);
        o0[g] += pg * v0; o1[g] += pg * v1;
      }
    }
    __syncthreads();
  }

  // per-wave partials
  if (lane == 0)
#pragma unroll
    for (int g = 0; g < GG; ++g) { red_m[w][g] = m_g[g]; red_l[w][g] = l_g[g]; }
#pragma unroll
  for (int g = 0; g < GG; ++g) {
    red_o[w][g][lane]      = o0[g];
    red_o[w][g][lane + 32] = o1[g];
  }
  // partial #8: the freshly appended token (not written to the cache)
  if (tid < GG) {
    float sE = 0.f;
    for (int d = 0; d < HD; ++d) sE += q_s[tid][d] * kn[d];
    red_m[8][tid] = sE; red_l[8][tid] = 1.f;
  }
  {
    const int g = tid >> 6, d = tid & 63;
    red_o[8][g][d] = vn[d];
  }
  __syncthreads();

  // combine 9 partials; thread <-> (g,d)
  const int g = tid >> 6, d = tid & 63;
  float mM = -__builtin_inff();
#pragma unroll
  for (int p = 0; p < 9; ++p) mM = fmaxf(mM, red_m[p][g]);
  float den = 0.f, acc = 0.f;
#pragma unroll
  for (int p = 0; p < 9; ++p) {
    const float e = __expf(red_m[p][g] - mM);
    den += red_l[p][g] * e;
    acc += red_o[p][g][d] * e;
  }
  o_out[(size_t)b * DD + (h * GG + g) * HD + d] = (bf16_t)(acc / den);
}

// =====================================================================
extern "C" void kernel_launch(void* const* d_in, const int* in_sizes, int n_in,
                              void* d_out, int out_size, void* d_ws, size_t ws_size,
                              hipStream_t stream) {
  (void)in_sizes; (void)n_in; (void)out_size; (void)ws_size;
  const float* x        = (const float*)d_in[0];
  const float* cache_k  = (const float*)d_in[1];
  const float* cache_v  = (const float*)d_in[2];
  const float* rms_w    = (const float*)d_in[3];
  const float* Wq       = (const float*)d_in[4];
  const float* Wk       = (const float*)d_in[5];
  const float* Wv       = (const float*)d_in[6];
  const float* Wo       = (const float*)d_in[7];
  const int*   ctx_lens = (const int*)d_in[8];
  float* out = (float*)d_out;

  char* ws = (char*)d_ws;
  bf16_t* h    = (bf16_t*)ws;                                  // 256 KB
  float*  qkv  = (float*)(ws + (size_t)BB * DD * 2);           // 768 KB
  bf16_t* oB   = (bf16_t*)(ws + (size_t)BB * DD * 2 + (size_t)BB * NQKV * 4);

  rmsnorm_kernel<<<BB, 256, 0, stream>>>(x, rms_w, h);
  qkv_gemm_kernel<<<dim3(NQKV / 16, BB / 16), 32, 0, stream>>>(h, Wq, Wk, Wv, qkv);
  rope_kernel<<<dim3(HQ + HKV, BB), 32, 0, stream>>>(qkv, ctx_lens);
  attn_kernel<<<dim3(HKV, BB), 256, 0, stream>>>(qkv, cache_k, cache_v, ctx_lens, oB);
  oproj_gemm_kernel<<<dim3(DD / 16, BB / 16), 32, 0, stream>>>(oB, Wo, x, out);
}
